// Model_78443282694273
// MI455X (gfx1250) — matrix operand order repaired and checked by reading, not compiled, not run on hardware
//
#include <hip/hip_runtime.h>
#include <hip/hip_bf16.h>

typedef __attribute__((ext_vector_type(16))) __bf16 v16bf;
typedef __attribute__((ext_vector_type(8)))  float  v8f;

#define S_LEN 2048
#define BATCH 32
#define DEMB  512
#define DRNN  256
#define G3    768   // 3*DRNN

union FragBF { uint4 q[2]; v16bf v; };

static __device__ __forceinline__ float sigmoidf_(float x) {
    return 1.0f / (1.0f + __expf(-x));
}

// ---------------------------------------------------------------------------
// K1: f32 -> bf16 conversion (grid-stride)
// ---------------------------------------------------------------------------
__global__ void cvt_f32_bf16(const float* __restrict__ src,
                             __bf16* __restrict__ dst, unsigned n) {
    unsigned i = blockIdx.x * blockDim.x + threadIdx.x;
    unsigned stride = gridDim.x * blockDim.x;
    for (; i < n; i += stride) dst[i] = (__bf16)src[i];
}

// ---------------------------------------------------------------------------
// K2: gi = X(65536x512) * W^T (W stored G x K row-major) + bias, bf16 out.
// Block 256 thr = 8 waves, tile M=128 x N=64, each wave 2x2 16x16 tiles.
// ---------------------------------------------------------------------------
__global__ __launch_bounds__(256) void gi_gemm(
    const __bf16* __restrict__ X, const __bf16* __restrict__ W,
    const float* __restrict__ bias, __bf16* __restrict__ gi)
{
    const int wave = threadIdx.x >> 5;
    const int lane = threadIdx.x & 31;
    const int hi   = lane >> 4;     // 0: lanes 0-15, 1: lanes 16-31
    const int l16  = lane & 15;
    const int mW = blockIdx.x * 128 + (wave >> 1) * 32;
    const int nW = blockIdx.y * 64  + (wave & 1) * 32;

    v8f c00 = {}, c01 = {}, c10 = {}, c11 = {};

    const __bf16* Xr0 = X + (size_t)(mW + l16) * DEMB;
    const __bf16* Xr1 = X + (size_t)(mW + 16 + l16) * DEMB;
    const __bf16* Wr0 = W + (size_t)(nW + l16) * DEMB;
    const __bf16* Wr1 = W + (size_t)(nW + 16 + l16) * DEMB;

    #pragma unroll 4
    for (int kk = 0; kk < DEMB; kk += 32) {
        FragBF a0, a1, b0, b1;
        const int ka = kk + hi * 8;   // A: elems 0-7 -> K=ka..ka+7, 8-15 -> +16
        a0.q[0] = *(const uint4*)(Xr0 + ka);
        a0.q[1] = *(const uint4*)(Xr0 + ka + 16);
        a1.q[0] = *(const uint4*)(Xr1 + ka);
        a1.q[1] = *(const uint4*)(Xr1 + ka + 16);
        const int kb = kk + hi * 8;   // B: elems 0-7 -> K=kb..kb+7, 8-15 -> +16
        b0.q[0] = *(const uint4*)(Wr0 + kb);
        b0.q[1] = *(const uint4*)(Wr0 + kb + 16);
        b1.q[0] = *(const uint4*)(Wr1 + kb);
        b1.q[1] = *(const uint4*)(Wr1 + kb + 16);
        c00 = __builtin_amdgcn_wmma_f32_16x16x32_bf16(false, a0.v, false, b0.v, (short)0, c00, false, false);
        c01 = __builtin_amdgcn_wmma_f32_16x16x32_bf16(false, a0.v, false, b1.v, (short)0, c01, false, false);
        c10 = __builtin_amdgcn_wmma_f32_16x16x32_bf16(false, a1.v, false, b0.v, (short)0, c10, false, false);
        c11 = __builtin_amdgcn_wmma_f32_16x16x32_bf16(false, a1.v, false, b1.v, (short)0, c11, false, false);
    }

    const float bv0 = bias[nW + l16];
    const float bv1 = bias[nW + 16 + l16];
    #pragma unroll
    for (int r = 0; r < 8; ++r) {
        const size_t row0 = (size_t)(mW + r + hi * 8);
        const size_t row1 = (size_t)(mW + 16 + r + hi * 8);
        __builtin_nontemporal_store((__bf16)(c00[r] + bv0), gi + row0 * G3 + nW + l16);
        __builtin_nontemporal_store((__bf16)(c01[r] + bv1), gi + row0 * G3 + nW + 16 + l16);
        __builtin_nontemporal_store((__bf16)(c10[r] + bv0), gi + row1 * G3 + nW + l16);
        __builtin_nontemporal_store((__bf16)(c11[r] + bv1), gi + row1 * G3 + nW + 16 + l16);
    }
}

// ---------------------------------------------------------------------------
// K3: sequential GRU scan. grid = 2 blocks (dir), 1024 threads = 32 waves.
// h lives in LDS (f32 + bf16 mirror). Per step: gh = h @ whh^T via WMMA
// (A-frags from LDS, B-frags streamed from L2-resident whh), gates in f32.
// gi slabs (48KB/step, bf16) are double-buffered in LDS via CDNA5 async
// global->LDS copies overlapped with the WMMA stage of the previous step.
// ---------------------------------------------------------------------------
__global__ __launch_bounds__(1024) void gru_scan(
    const __bf16* __restrict__ whh0, const __bf16* __restrict__ whh1,
    const float* __restrict__ bhh0,  const float* __restrict__ bhh1,
    const __bf16* __restrict__ gi0,  const __bf16* __restrict__ gi1,
    float* __restrict__ out)
{
    extern __shared__ char smem[];
    float*  sh_gh = (float*)smem;                                       // 32x768 f32  (96KB)
    float*  sh_h  = (float*)(smem + BATCH * G3 * 4);                    // 32x256 f32  (32KB)
    __bf16* sh_hb = (__bf16*)(smem + BATCH * G3 * 4 + BATCH * DRNN * 4);// 32x256 bf16 (16KB)
    __bf16* sh_gi = (__bf16*)(smem + BATCH * G3 * 4 + BATCH * DRNN * 6);// 2x 32x768 bf16 (96KB)

    const int dir = blockIdx.x;
    const __bf16* whh = dir ? whh1 : whh0;
    const float*  bhh = dir ? bhh1 : bhh0;
    const __bf16* gi  = dir ? gi1  : gi0;

    const int tid  = threadIdx.x;
    const int wave = tid >> 5, lane = tid & 31;
    const int hi   = lane >> 4, l16 = lane & 15;
    const int mTile = wave >> 4;            // 0..1 (rows 0-15 / 16-31)
    const int nTB   = (wave & 15) * 3;      // 3 n-tiles per wave -> 48 tiles

    for (int i = tid; i < BATCH * DRNN; i += 1024) {
        sh_h[i]  = 0.0f;
        sh_hb[i] = (__bf16)0.0f;
    }

    // Issue async copy of the first timestep's gi slab into buffer 0.
    // Each thread moves 3 x 16B; 1024 threads cover the 48KB slab.
    {
        const int tt0 = dir ? (S_LEN - 1) : 0;
        const char* src = (const char*)(gi + (size_t)tt0 * BATCH * G3);
        const unsigned ldsbase = (unsigned)(uintptr_t)sh_gi;
        #pragma unroll
        for (int r = 0; r < 3; ++r) {
            const unsigned off = (unsigned)tid * 16u + (unsigned)r * 16384u;
            const unsigned ldsa = ldsbase + off;
            const void* g = src + off;
            asm volatile("global_load_async_to_lds_b128 %0, %1, off"
                         :: "v"(ldsa), "v"(g) : "memory");
        }
    }
    __syncthreads();

    const __bf16* hbRow = sh_hb + (mTile * 16 + l16) * DRNN;
    const __bf16* wRow0 = whh + (size_t)((nTB + 0) * 16 + l16) * DRNN;
    const __bf16* wRow1 = whh + (size_t)((nTB + 1) * 16 + l16) * DRNN;
    const __bf16* wRow2 = whh + (size_t)((nTB + 2) * 16 + l16) * DRNN;

    for (int t = 0; t < S_LEN; ++t) {
        const int tt = dir ? (S_LEN - 1 - t) : t;

        // Kick off async copy of the *next* step's gi slab into the other
        // buffer; it overlaps with this step's WMMA stage.
        const bool have_next = (t + 1 < S_LEN);
        if (have_next) {
            const int ttn = dir ? (tt - 1) : (tt + 1);
            const char* src = (const char*)(gi + (size_t)ttn * BATCH * G3);
            const unsigned ldsbase =
                (unsigned)(uintptr_t)(sh_gi + (size_t)((t + 1) & 1) * BATCH * G3);
            #pragma unroll
            for (int r = 0; r < 3; ++r) {
                const unsigned off = (unsigned)tid * 16u + (unsigned)r * 16384u;
                const unsigned ldsa = ldsbase + off;
                const void* g = src + off;
                asm volatile("global_load_async_to_lds_b128 %0, %1, off"
                             :: "v"(ldsa), "v"(g) : "memory");
            }
        }

        // gh = h @ whh^T  (M=32, N=768, K=256), 3 accum tiles per wave.
        v8f c0 = {}, c1 = {}, c2 = {};
        #pragma unroll
        for (int k = 0; k < DRNN; k += 32) {
            FragBF a, b;
            const int ka = k + hi * 8;
            a.q[0] = *(const uint4*)(hbRow + ka);          // ds_load_b128
            a.q[1] = *(const uint4*)(hbRow + ka + 16);
            const int kb = k + hi * 8;
            b.q[0] = *(const uint4*)(wRow0 + kb);
            b.q[1] = *(const uint4*)(wRow0 + kb + 16);
            c0 = __builtin_amdgcn_wmma_f32_16x16x32_bf16(false, a.v, false, b.v, (short)0, c0, false, false);
            b.q[0] = *(const uint4*)(wRow1 + kb);
            b.q[1] = *(const uint4*)(wRow1 + kb + 16);
            c1 = __builtin_amdgcn_wmma_f32_16x16x32_bf16(false, a.v, false, b.v, (short)0, c1, false, false);
            b.q[0] = *(const uint4*)(wRow2 + kb);
            b.q[1] = *(const uint4*)(wRow2 + kb + 16);
            c2 = __builtin_amdgcn_wmma_f32_16x16x32_bf16(false, a.v, false, b.v, (short)0, c2, false, false);
        }

        // Stage gh tiles into LDS for the gate stage.
        #pragma unroll
        for (int r = 0; r < 8; ++r) {
            const int row = mTile * 16 + r + hi * 8;
            sh_gh[row * G3 + (nTB + 0) * 16 + l16] = c0[r];
            sh_gh[row * G3 + (nTB + 1) * 16 + l16] = c1[r];
            sh_gh[row * G3 + (nTB + 2) * 16 + l16] = c2[r];
        }

        // Ensure this step's gi slab has landed in LDS. Async loads complete
        // in order: if we just issued 3 more (for t+1), waiting down to <=3
        // guarantees the 3 older (step-t) ones are done.
        if (have_next) {
            asm volatile("s_wait_asynccnt 0x3" ::: "memory");
        } else {
            asm volatile("s_wait_asynccnt 0x0" ::: "memory");
        }
        __syncthreads();

        // Gates: 32*256 = 8192 (b,j) elements, 8 per thread.
        const __bf16* giT = sh_gi + (size_t)(t & 1) * BATCH * G3;
        #pragma unroll
        for (int i = 0; i < 8; ++i) {
            const int idx = i * 1024 + tid;
            const int bb = idx >> 8, j = idx & 255;
            const float gr = sh_gh[bb * G3 + j]        + bhh[j];
            const float gz = sh_gh[bb * G3 + 256 + j]  + bhh[256 + j];
            const float gn = sh_gh[bb * G3 + 512 + j]  + bhh[512 + j];
            const float ir = (float)giT[bb * G3 + j];
            const float iz = (float)giT[bb * G3 + 256 + j];
            const float in = (float)giT[bb * G3 + 512 + j];
            const float rg = sigmoidf_(ir + gr);
            const float zg = sigmoidf_(iz + gz);
            const float ng = tanhf(in + rg * gn);
            const float hprev = sh_h[bb * DRNN + j];
            const float hy = ng + zg * (hprev - ng);
            sh_h[bb * DRNN + j]  = hy;
            sh_hb[bb * DRNN + j] = (__bf16)hy;
            out[((size_t)tt * BATCH + bb) * (2 * DRNN) + dir * DRNN + j] = hy;
        }
        __syncthreads();
    }

    // Final hidden state: hidden[dir][b][j] appended after the output tensor.
    for (int i = tid; i < BATCH * DRNN; i += 1024)
        out[(size_t)S_LEN * BATCH * 2 * DRNN + dir * BATCH * DRNN + i] = sh_h[i];
}

// ---------------------------------------------------------------------------
extern "C" void kernel_launch(void* const* d_in, const int* in_sizes, int n_in,
                              void* d_out, int out_size, void* d_ws, size_t ws_size,
                              hipStream_t stream) {
    const float* x      = (const float*)d_in[0];
    const float* w_ih_f = (const float*)d_in[1];
    const float* w_hh_f = (const float*)d_in[2];
    const float* b_ih_f = (const float*)d_in[3];
    const float* b_hh_f = (const float*)d_in[4];
    const float* w_ih_b = (const float*)d_in[5];
    const float* w_hh_b = (const float*)d_in[6];
    const float* b_ih_b = (const float*)d_in[7];
    const float* b_hh_b = (const float*)d_in[8];
    float* out = (float*)d_out;
    char*  ws  = (char*)d_ws;

    const size_t NX = (size_t)S_LEN * BATCH * DEMB;        // 33.5M
    __bf16* xbf  = (__bf16*)ws;
    __bf16* wihf = (__bf16*)(ws + NX * 2);
    __bf16* wihb = wihf + (size_t)G3 * DEMB;
    __bf16* whhf = wihb + (size_t)G3 * DEMB;
    __bf16* whhb = whhf + (size_t)G3 * DRNN;
    __bf16* gif  = whhb + (size_t)G3 * DRNN;
    __bf16* gib  = gif + (size_t)S_LEN * BATCH * G3;

    cvt_f32_bf16<<<2048, 256, 0, stream>>>(x,      xbf,  (unsigned)NX);
    cvt_f32_bf16<<<256,  256, 0, stream>>>(w_ih_f, wihf, G3 * DEMB);
    cvt_f32_bf16<<<256,  256, 0, stream>>>(w_ih_b, wihb, G3 * DEMB);
    cvt_f32_bf16<<<128,  256, 0, stream>>>(w_hh_f, whhf, G3 * DRNN);
    cvt_f32_bf16<<<128,  256, 0, stream>>>(w_hh_b, whhb, G3 * DRNN);

    dim3 g2(512, 12, 1);   // M=65536/128, N=768/64
    gi_gemm<<<g2, 256, 0, stream>>>(xbf, wihf, b_ih_f, gif);
    gi_gemm<<<g2, 256, 0, stream>>>(xbf, wihb, b_ih_b, gib);

    const size_t shbytes = BATCH * G3 * 4      // sh_gh
                         + BATCH * DRNN * 4    // sh_h
                         + BATCH * DRNN * 2    // sh_hb
                         + 2 * BATCH * G3 * 2; // sh_gi double buffer  -> 240KB
    gru_scan<<<2, 1024, shbytes, stream>>>(whhf, whhb, b_hh_f, b_hh_b, gif, gib, out);
}
